// RNN_AE_31370441129998
// MI455X (gfx1250) — matrix operand order repaired and checked by reading, not compiled, not run on hardware
//
#include <hip/hip_runtime.h>

typedef __attribute__((ext_vector_type(16))) _Float16 v16h;
typedef __attribute__((ext_vector_type(8)))  _Float16 v8h;
typedef __attribute__((ext_vector_type(8)))  float    v8f;
typedef __attribute__((ext_vector_type(2)))  float    v2f;

#define GLB __attribute__((address_space(1)))

#define HS   136            // padded LDS row stride in halves (conflict-free 16B loads)
#define SLOT (16 * HS)      // one 16x128 tile slot (halves)
#define NW   2              // waves per block
#define ROWS_PER_BLOCK (NW * 16)
#define B_TOT 65536
#define T_SEQ 12

// ---- opaque pointer barriers: defeat LICM of loop-invariant weight loads ----
// Keeps the pointer in SGPRs but makes its value unknown per call site, so loads
// cannot be hoisted across time steps. Returned as address_space(1) so accesses
// lower to global_load (LOADcnt only) instead of flat_load (LOADcnt+DScnt).
__device__ __forceinline__ GLB const _Float16* opqh(const _Float16* p) {
  unsigned long long u = (unsigned long long)p;
  asm volatile("" : "+s"(u));
  return (GLB const _Float16*)u;
}
__device__ __forceinline__ GLB const float* opqf(const float* p) {
  unsigned long long u = (unsigned long long)p;
  asm volatile("" : "+s"(u));
  return (GLB const float*)u;
}
__device__ __forceinline__ GLB const v2f* opq2(const v2f* p) {
  unsigned long long u = (unsigned long long)p;
  asm volatile("" : "+s"(u));
  return (GLB const v2f*)u;
}

__device__ __forceinline__ v8f wmma_f16(v16h a, v16h b, v8f c) {
  return __builtin_amdgcn_wmma_f32_16x16x32_f16(false, a, false, b, (short)0, c, false, false);
}
// A fragment (16x32 f16, MxK): lane: row = lane%16; halves 0..7 -> K = kc*32 + 8*(lane/16)+i,
// halves 8..15 -> K = kc*32 + 16 + 8*(lane/16)+i. Two 16B LDS loads.
__device__ __forceinline__ v16h ldsA(const _Float16* s, int kc, int m, int hi) {
  const _Float16* p = s + m * HS + kc * 32 + hi * 8;
  v8h lo = *(const v8h*)p;
  v8h up = *(const v8h*)(p + 16);
  return __builtin_shufflevector(lo, up, 0, 1, 2, 3, 4, 5, 6, 7, 8, 9, 10, 11, 12, 13, 14, 15);
}
__device__ __forceinline__ void loadFrags(const _Float16* s, v16h a[4], int n, int hi) {
#pragma unroll
  for (int kc = 0; kc < 4; kc++) a[kc] = ldsA(s, kc, n, hi);
}
// B fragment (32x16 f16, KxN): lane: col = lane%16 = W row; halves 0..7 -> K = kc*32 + 8*(lane/16)+i,
// halves 8..15 -> K = kc*32 + 16 + 8*(lane/16)+i. Two 16B global reads.
__device__ __forceinline__ v16h gB(GLB const _Float16* W, int row, int kc, int hi) {
  GLB const _Float16* p = W + row * 128 + kc * 32 + hi * 8;
  v8h lo = *(GLB const v8h*)p;
  v8h up = *(GLB const v8h*)(p + 16);
  return __builtin_shufflevector(lo, up, 0, 1, 2, 3, 4, 5, 6, 7, 8, 9, 10, 11, 12, 13, 14, 15);
}
__device__ __forceinline__ float sigm(float x) { return 1.f / (1.f + __expf(-x)); }
__device__ __forceinline__ float tanha(float x) {
  float e = __expf(-2.f * x);
  return (1.f - e) / (1.f + e);
}

// One GRU cell for this wave's 16 rows. hbuf updated in place (old-h A-frags hoisted
// to registers first; per-column old-h read happens before that column's write).
template <bool HZ>
__device__ __forceinline__ void gru_cell(_Float16* hbuf, const v16h aIn[4],
                                         const _Float16* Wih_, const _Float16* Whh_,
                                         const float* bih_, const float* bhh_,
                                         int n, int hi) {
  GLB const _Float16* Wih = opqh(Wih_);
  GLB const _Float16* Whh = opqh(Whh_);
  GLB const float* bih = opqf(bih_);
  GLB const float* bhh = opqf(bhh_);
  v16h aH[4];
  if (!HZ) loadFrags(hbuf, aH, n, hi);
#pragma clang loop unroll(disable)
  for (int nc = 0; nc < 8; nc++) {
    int j = nc * 16 + n;
    // accumulators start from inline-zero C; biases folded into the tail
    v8f aR = {}, aZ = {}, aNX = {}, aNH = {};
#pragma unroll
    for (int kc = 0; kc < 4; kc++) {
      aR  = wmma_f16(aIn[kc], gB(Wih, j,       kc, hi), aR);
      aZ  = wmma_f16(aIn[kc], gB(Wih, 128 + j, kc, hi), aZ);
      aNX = wmma_f16(aIn[kc], gB(Wih, 256 + j, kc, hi), aNX);
    }
    if (!HZ) {
#pragma unroll
      for (int kc = 0; kc < 4; kc++) {
        aR  = wmma_f16(aH[kc], gB(Whh, j,       kc, hi), aR);
        aZ  = wmma_f16(aH[kc], gB(Whh, 128 + j, kc, hi), aZ);
        aNH = wmma_f16(aH[kc], gB(Whh, 256 + j, kc, hi), aNH);
      }
    }
    float brz = bih[j] + bhh[j];
    float bzz = bih[128 + j] + bhh[128 + j];
    float bnx = bih[256 + j];
    float bnh = bhh[256 + j];
#pragma unroll
    for (int v = 0; v < 8; v++) {
      int m = v + 8 * hi;  // D layout: lane holds col j; VGPR v -> row v + 8*(lane/16)
      float r  = sigm(aR[v] + brz);
      float z  = sigm(aZ[v] + bzz);
      float nn = tanha(aNX[v] + bnx + r * (aNH[v] + bnh));
      float ho = HZ ? 0.f : (float)hbuf[m * HS + j];
      hbuf[m * HS + j] = (_Float16)((1.f - z) * nn + z * ho);
    }
  }
}

// dst = act(A @ W^T + b), A given as register fragments, dst written to LDS tile.
template <bool RELU>
__device__ __forceinline__ void gemm128(const v16h aA[4], const _Float16* W_,
                                        const float* b_, _Float16* dst, int n, int hi) {
  GLB const _Float16* W = opqh(W_);
  GLB const float* bb = opqf(b_);
#pragma clang loop unroll(disable)
  for (int nc = 0; nc < 8; nc++) {
    int j = nc * 16 + n;
    v8f acc = {};
#pragma unroll
    for (int kc = 0; kc < 4; kc++) acc = wmma_f16(aA[kc], gB(W, j, kc, hi), acc);
    float bj = bb[j];
#pragma unroll
    for (int v = 0; v < 8; v++) {
      float x = acc[v] + bj;
      if (RELU) x = fmaxf(x, 0.f);
      dst[(v + 8 * hi) * HS + j] = (_Float16)x;
    }
  }
}

// Embedding for step s directly into A-fragment layout: relu(x_true[:,s+1] @ Wemb^T + bemb)
__device__ __forceinline__ void build_xf(v16h xf[4], const v2f* xt2_, const v2f* we2_,
                                         const float* bemb_, int row, int s, int hi) {
  GLB const v2f* xt2 = opq2(xt2_);
  GLB const v2f* we2 = opq2(we2_);
  GLB const float* bemb = opqf(bemb_);
  v2f xv = xt2[row * T_SEQ + (s + 1)];
#pragma unroll
  for (int kc = 0; kc < 4; kc++) {
#pragma unroll
    for (int part = 0; part < 2; part++) {
#pragma unroll
      for (int i = 0; i < 8; i++) {
        int e = kc * 32 + part * 16 + hi * 8 + i;
        v2f w = we2[e];
        float vv = xv.x * w.x + xv.y * w.y + bemb[e];
        xf[kc][part * 8 + i] = (_Float16)fmaxf(vv, 0.f);
      }
    }
  }
}

// y(t) = h2 @ Wout^T + bout : lane -> (batch row = n, out dim = hi)
__device__ __forceinline__ void emit_y(const _Float16* h2, const float* Wout_,
                                       const float* bout_, float* out, int bbase,
                                       int t, int n, int hi) {
  GLB const float* Wout = opqf(Wout_);
  GLB const float* bout = opqf(bout_);
  float acc = bout[hi];
  const _Float16* hrow = h2 + n * HS;
#pragma unroll
  for (int k8 = 0; k8 < 16; k8++) {
    v8h hv = *(const v8h*)(hrow + k8 * 8);
#pragma unroll
    for (int i = 0; i < 8; i++) acc += (float)hv[i] * Wout[hi * 128 + k8 * 8 + i];
  }
  ((GLB float*)out)[(bbase + n) * T_SEQ * 2 + t * 2 + hi] = acc;
}

__global__ __launch_bounds__(NW * 32) void rnn_ae_fused(
    const float* __restrict__ x_true,
    const float* __restrict__ enc_Wemb, const float* __restrict__ enc_bemb,
    const float* __restrict__ enc_bih, const float* __restrict__ enc_bhh,
    const float* __restrict__ enc_bt, const float* __restrict__ dec_bin,
    const float* __restrict__ dec_bih, const float* __restrict__ dec_bhh,
    const float* __restrict__ dec_Wout, const float* __restrict__ dec_bout,
    const _Float16* __restrict__ eWih, const _Float16* __restrict__ eWhh,
    const _Float16* __restrict__ dWih, const _Float16* __restrict__ dWhh,
    const _Float16* __restrict__ wWt, const _Float16* __restrict__ wWin,
    float* __restrict__ out) {
  __shared__ _Float16 smem[NW * 4 * SLOT];
  const int lane = threadIdx.x & 31;
  const int wave = threadIdx.x >> 5;
  const int n = lane & 15, hi = lane >> 4;
  const int bbase = blockIdx.x * ROWS_PER_BLOCK + wave * 16;

  _Float16* h0  = smem + wave * 4 * SLOT;
  _Float16* h1  = h0 + SLOT;
  _Float16* h2  = h1 + SLOT;
  _Float16* inb = h2 + SLOT;

  const int row = bbase + n;  // this lane's batch row for A-frags / output
  const v2f* xt2 = (const v2f*)x_true;    // [B][T] of float pairs
  const v2f* we2 = (const v2f*)enc_Wemb;  // [128] of float pairs

  v16h xf[4], aL[4];

  // ---------------- encoder (steps use x_true[:, 1..10]) ----------------
  // step 0 peeled: hidden is zero
  build_xf(xf, xt2, we2, enc_bemb, row, 0, hi);
  gru_cell<true>(h0, xf, eWih, eWhh, enc_bih, enc_bhh, n, hi);
  loadFrags(h0, aL, n, hi);
  gru_cell<true>(h1, aL, eWih + 49152, eWhh + 49152, enc_bih + 384, enc_bhh + 384, n, hi);
  loadFrags(h1, aL, n, hi);
  gru_cell<true>(h2, aL, eWih + 98304, eWhh + 98304, enc_bih + 768, enc_bhh + 768, n, hi);

#pragma clang loop unroll(disable)
  for (int s = 1; s < 10; s++) {
    build_xf(xf, xt2, we2, enc_bemb, row, s, hi);
    gru_cell<false>(h0, xf, eWih, eWhh, enc_bih, enc_bhh, n, hi);
    loadFrags(h0, aL, n, hi);
    gru_cell<false>(h1, aL, eWih + 49152, eWhh + 49152, enc_bih + 384, enc_bhh + 384, n, hi);
    loadFrags(h1, aL, n, hi);
    gru_cell<false>(h2, aL, eWih + 98304, eWhh + 98304, enc_bih + 768, enc_bhh + 768, n, hi);
  }

  // rep = alpha @ Wt^T + bt (alpha = h2) -> stored into h0 slot (decoder step-0 input)
  loadFrags(h2, aL, n, hi);
  gemm128<false>(aL, wWt, enc_bt, h0, n, hi);

  // ---------------- decoder (12 steps) ----------------
  // step 0 peeled: src = rep (in h0), hidden is zero (slots hold stale encoder data)
  loadFrags(h0, aL, n, hi);
  gemm128<true>(aL, wWin, dec_bin, inb, n, hi);  // inb = relu(rep @ Win^T + bin)
  loadFrags(inb, aL, n, hi);
  gru_cell<true>(h0, aL, dWih, dWhh, dec_bih, dec_bhh, n, hi);
  loadFrags(h0, aL, n, hi);
  gru_cell<true>(h1, aL, dWih + 49152, dWhh + 49152, dec_bih + 384, dec_bhh + 384, n, hi);
  loadFrags(h1, aL, n, hi);
  gru_cell<true>(h2, aL, dWih + 98304, dWhh + 98304, dec_bih + 768, dec_bhh + 768, n, hi);
  emit_y(h2, dec_Wout, dec_bout, out, bbase, 0, n, hi);

#pragma clang loop unroll(disable)
  for (int t = 1; t < 12; t++) {
    loadFrags(h2, aL, n, hi);                      // carry r = top-layer hidden
    gemm128<true>(aL, wWin, dec_bin, inb, n, hi);  // inb = relu(r @ Win^T + bin)
    loadFrags(inb, aL, n, hi);
    gru_cell<false>(h0, aL, dWih, dWhh, dec_bih, dec_bhh, n, hi);
    loadFrags(h0, aL, n, hi);
    gru_cell<false>(h1, aL, dWih + 49152, dWhh + 49152, dec_bih + 384, dec_bhh + 384, n, hi);
    loadFrags(h1, aL, n, hi);
    gru_cell<false>(h2, aL, dWih + 98304, dWhh + 98304, dec_bih + 768, dec_bhh + 768, n, hi);
    emit_y(h2, dec_Wout, dec_bout, out, bbase, t, n, hi);
  }
}

__global__ void cvt_f32_f16(const float* __restrict__ s, _Float16* __restrict__ d, int nElem) {
  int i = blockIdx.x * blockDim.x + threadIdx.x;
  if (i < nElem) d[i] = (_Float16)s[i];
}

extern "C" void kernel_launch(void* const* d_in, const int* in_sizes, int n_in,
                              void* d_out, int out_size, void* d_ws, size_t ws_size,
                              hipStream_t stream) {
  const float* x_true   = (const float*)d_in[0];
  const float* enc_Wemb = (const float*)d_in[1];
  const float* enc_bemb = (const float*)d_in[2];
  const float* enc_Wih  = (const float*)d_in[3];
  const float* enc_Whh  = (const float*)d_in[4];
  const float* enc_bih  = (const float*)d_in[5];
  const float* enc_bhh  = (const float*)d_in[6];
  const float* enc_Wt   = (const float*)d_in[7];
  const float* enc_bt   = (const float*)d_in[8];
  const float* dec_Win  = (const float*)d_in[9];
  const float* dec_bin  = (const float*)d_in[10];
  const float* dec_Wih  = (const float*)d_in[11];
  const float* dec_Whh  = (const float*)d_in[12];
  const float* dec_bih  = (const float*)d_in[13];
  const float* dec_bhh  = (const float*)d_in[14];
  const float* dec_Wout = (const float*)d_in[15];
  const float* dec_bout = (const float*)d_in[16];

  _Float16* ws   = (_Float16*)d_ws;
  _Float16* eWih = ws;
  _Float16* eWhh = eWih + 147456;
  _Float16* dWih = eWhh + 147456;
  _Float16* dWhh = dWih + 147456;
  _Float16* wWt  = dWhh + 147456;
  _Float16* wWin = wWt + 16384;

  cvt_f32_f16<<<(147456 + 255) / 256, 256, 0, stream>>>(enc_Wih, eWih, 147456);
  cvt_f32_f16<<<(147456 + 255) / 256, 256, 0, stream>>>(enc_Whh, eWhh, 147456);
  cvt_f32_f16<<<(147456 + 255) / 256, 256, 0, stream>>>(dec_Wih, dWih, 147456);
  cvt_f32_f16<<<(147456 + 255) / 256, 256, 0, stream>>>(dec_Whh, dWhh, 147456);
  cvt_f32_f16<<<(16384 + 255) / 256, 256, 0, stream>>>(enc_Wt, wWt, 16384);
  cvt_f32_f16<<<(16384 + 255) / 256, 256, 0, stream>>>(dec_Win, wWin, 16384);

  dim3 grid(B_TOT / ROWS_PER_BLOCK);
  rnn_ae_fused<<<grid, NW * 32, 0, stream>>>(
      x_true, enc_Wemb, enc_bemb, enc_bih, enc_bhh, enc_bt, dec_bin, dec_bih,
      dec_bhh, dec_Wout, dec_bout, eWih, eWhh, dWih, dWhh, wWt, wWin,
      (float*)d_out);
}
